// MoEMLP_53395033424578
// MI455X (gfx1250) — matrix operand order repaired and checked by reading, not compiled, not run on hardware
//
#include <hip/hip_runtime.h>
#include <hip/hip_bf16.h>
#include <math.h>

// MoE expert-choice MLP for MI455X (gfx1250, wave32).
// Grouped GEMMs run as bf16 WMMA with f32 accumulation; tiles are staged into
// LDS by the Tensor Data Mover (tensor_load_to_lds) with hardware LDS padding.

typedef __bf16 bf16;
typedef bf16  v16bf __attribute__((ext_vector_type(16)));
typedef float v8f   __attribute__((ext_vector_type(8)));
typedef unsigned int v4u __attribute__((ext_vector_type(4)));
typedef int   v8i   __attribute__((ext_vector_type(8)));
typedef int   v4i   __attribute__((ext_vector_type(4)));

#if __has_builtin(__builtin_amdgcn_tensor_load_to_lds)
#define USE_TDM 1
#endif

#define B_ 4
#define S_ 2048
#define H_ 1024
#define E_ 8
#define K_ 512
#define F_ 4096

// ---------------- GEMM tiling ----------------
#define BM 128
#define BN 128
#define BK 64    // two 16x16x32 WMMA sub-steps per staged tile
#define LDT 72   // padded LDS row stride in bf16: 128B data + 16B pad = 144B

union FragAB { v16bf v; uint4 q[2]; };
union FragC  { v8f  v; float f[8]; };

__device__ __forceinline__ uint32_t lds_off32(const void* p) {
  // flat LDS aperture keeps the byte offset in addr[31:0]
  return (uint32_t)(uintptr_t)p;
}

#ifdef USE_TDM
// Issue one TDM 2D tile load: [128 rows x 64 bf16] from row-major global
// (row stride ld_elems) into LDS at lds_byte with 16B pad every 128B row.
__device__ __forceinline__ void tdm_load_tile(uint32_t lds_byte, const bf16* gptr,
                                              uint32_t ld_elems) {
  uint64_t ga = (uint64_t)(uintptr_t)gptr;
  v4u g0;
  g0[0] = 1u;                                   // count=1, user descriptor
  g0[1] = lds_byte;                             // lds_addr (bytes)
  g0[2] = (uint32_t)ga;                         // global_addr[31:0]
  g0[3] = ((uint32_t)(ga >> 32) & 0x01FFFFFFu)  // global_addr[56:32]
          | (2u << 30);                         // type=2 ("image")
  v8i g1;
  g1[0] = (int)((1u << 16)      // data_size=1 -> 2-byte elements
              | (1u << 20)      // pad_enable
              | (4u << 22)      // pad_interval=4 -> every 32 dwords (128B)
              | (3u << 25));    // pad_amount=3 -> 4 dwords (16B)
  g1[1] = (int)((ld_elems & 0xFFFFu) << 16);                  // tensor_dim0[15:0]
  g1[2] = (int)(((ld_elems >> 16) & 0xFFFFu) | (128u << 16)); // td0 hi | tensor_dim1=128
  g1[3] = (int)(64u << 16);                                   // td1 hi=0 | tile_dim0=64
  g1[4] = (int)(128u);                                        // tile_dim1=128, tile_dim2=0
  g1[5] = (int)ld_elems;                                      // tensor_dim0_stride[31:0]
  g1[6] = 0;                                                  // stride hi | dim1_stride lo
  g1[7] = 0;
  v4i z4 = {0, 0, 0, 0};
  v8i z8 = {0, 0, 0, 0, 0, 0, 0, 0};
  __builtin_amdgcn_tensor_load_to_lds(g0, g1, z4, z4, z8, 0);
}
#else
struct Stage { uint4 a[4]; uint4 b[4]; };
__device__ __forceinline__ void ld_global(Stage& st, const bf16* Ag, int lda,
                                          const bf16* Bg, int ldb, int kc, int tid) {
#pragma unroll
  for (int u = 0; u < 4; ++u) {
    int c = tid + u * 256, r = c >> 3, p = c & 7;
    st.a[u] = *(const uint4*)(Ag + (size_t)r * lda + kc + p * 8);
    st.b[u] = *(const uint4*)(Bg + (size_t)r * ldb + kc + p * 8);
  }
}
__device__ __forceinline__ void st_lds(const Stage& st, bf16* dA, bf16* dB, int tid) {
#pragma unroll
  for (int u = 0; u < 4; ++u) {
    int c = tid + u * 256, r = c >> 3, p = c & 7;
    *(uint4*)(dA + r * LDT + p * 8) = st.a[u];
    *(uint4*)(dB + r * LDT + p * 8) = st.b[u];
  }
}
#endif

// C[BM,BN] += A[BM,kred] * B[kred,BN]; B supplied column-major (ldb per column).
__device__ __forceinline__ void gemm_tile(const bf16* Ag, int lda,
                                          const bf16* Bg, int ldb, int kred, int tid,
                                          bf16 (*sA)[BM * LDT], bf16 (*sB)[BN * LDT],
                                          FragC acc[2][4]) {
  const int lane = tid & 31, wave = tid >> 5;
  const int wm = (wave >> 1) * 32;   // 4 waves along M
  const int wn = (wave & 1) * 64;    // 2 waves along N
  const int mr = lane & 15, grp = lane >> 4;
  const int nk = kred / BK;
  int buf = 0;

#ifdef USE_TDM
  if (wave == 0)      tdm_load_tile(lds_off32(&sA[0][0]), Ag, (uint32_t)lda);
  else if (wave == 1) tdm_load_tile(lds_off32(&sB[0][0]), Bg, (uint32_t)ldb);
  for (int k = 0; k < nk; ++k) {
    if (k + 1 < nk) {  // prev barrier guarantees buf^1 readers are done
      if (wave == 0)      tdm_load_tile(lds_off32(&sA[buf ^ 1][0]),
                                        Ag + (size_t)(k + 1) * BK, (uint32_t)lda);
      else if (wave == 1) tdm_load_tile(lds_off32(&sB[buf ^ 1][0]),
                                        Bg + (size_t)(k + 1) * BK, (uint32_t)ldb);
    }
    if (wave < 2) {
      if (k + 1 < nk) __builtin_amdgcn_s_wait_tensorcnt(1);  // tile k resident
      else            __builtin_amdgcn_s_wait_tensorcnt(0);
    }
    __syncthreads();   // tile k visible to all waves
#else
  Stage st;
  ld_global(st, Ag, lda, Bg, ldb, 0, tid);
  st_lds(st, sA[0], sB[0], tid);
  __syncthreads();
  for (int k = 0; k < nk; ++k) {
    if (k + 1 < nk) ld_global(st, Ag, lda, Bg, ldb, (k + 1) * BK, tid);
#endif
#pragma unroll
    for (int s = 0; s < 2; ++s) {   // two K=32 sub-steps of the 64-wide tile
      FragAB fa[2], fb[4];
#pragma unroll
      for (int i = 0; i < 2; ++i) {
        // A 16-bit 16x32 layout: lane m=lane&15, half g holds K=g*8..+7, 16+g*8..+7
        const bf16* p = &sA[buf][(wm + i * 16 + mr) * LDT + s * 32 + grp * 8];
        fa[i].q[0] = *(const uint4*)(p);
        fa[i].q[1] = *(const uint4*)(p + 16);
      }
#pragma unroll
      for (int j = 0; j < 4; ++j) {
        // B 16-bit 32x16 layout: lane n=lane&15, half g holds K=g*8..+7, 16+g*8..+7
        const bf16* p = &sB[buf][(wn + j * 16 + mr) * LDT + s * 32 + grp * 8];
        fb[j].q[0] = *(const uint4*)(p);
        fb[j].q[1] = *(const uint4*)(p + 16);
      }
#pragma unroll
      for (int i = 0; i < 2; ++i)
#pragma unroll
        for (int j = 0; j < 4; ++j)
          acc[i][j].v = __builtin_amdgcn_wmma_f32_16x16x32_bf16(
              false, fa[i].v, false, fb[j].v, (short)0, acc[i][j].v, false, false);
    }
#ifdef USE_TDM
    __syncthreads();   // all waves done reading buf before TDM overwrites it
#else
    if (k + 1 < nk) st_lds(st, sA[buf ^ 1], sB[buf ^ 1], tid);
    __syncthreads();
#endif
    buf ^= 1;
  }
}

// ---------------- small kernels ----------------
__global__ __launch_bounds__(256) void zero4(float* __restrict__ p) {
  size_t i = ((size_t)blockIdx.x * 256 + threadIdx.x) * 4;
  *(float4*)(p + i) = make_float4(0.f, 0.f, 0.f, 0.f);
}

// W1[E][H][F] f32 -> W1T[E][F][H] bf16 (linear write order e,f,h)
__global__ __launch_bounds__(256) void conv_w1t(const float* __restrict__ W1, bf16* __restrict__ W1T) {
  size_t t = (size_t)blockIdx.x * 256 + threadIdx.x;
  size_t e = t / ((size_t)F_ * H_), rem = t % ((size_t)F_ * H_);
  size_t f = rem / H_, h = rem % H_;
  W1T[t] = (bf16)W1[(e * H_ + h) * F_ + f];
}

// W2[E][F][H] f32 -> W2T[E][H][F] bf16 (linear write order e,h,f)
__global__ __launch_bounds__(256) void conv_w2t(const float* __restrict__ W2, bf16* __restrict__ W2T) {
  size_t t = (size_t)blockIdx.x * 256 + threadIdx.x;
  size_t e = t / ((size_t)H_ * F_), rem = t % ((size_t)H_ * F_);
  size_t h = rem / F_, f = rem % F_;
  W2T[t] = (bf16)W2[(e * F_ + f) * H_ + h];
}

// one wave per token: logits over E experts + softmax -> Sm[B][E][S]
__global__ __launch_bounds__(256) void gate_softmax(const float* __restrict__ x,
                                                    const float* __restrict__ Wg,
                                                    float* __restrict__ Sm) {
  const int wave = threadIdx.x >> 5, lane = threadIdx.x & 31;
  const int t = blockIdx.x * 8 + wave;   // token in [0, B*S)
  const float* xp = x + (size_t)t * H_;
  float acc[E_];
#pragma unroll
  for (int e = 0; e < E_; ++e) acc[e] = 0.f;
  for (int i = 0; i < H_ / 32; ++i) {
    int idx = i * 32 + lane;
    float xv = xp[idx];
#pragma unroll
    for (int e = 0; e < E_; ++e) acc[e] += xv * Wg[e * H_ + idx];
  }
#pragma unroll
  for (int e = 0; e < E_; ++e)
    for (int off = 16; off > 0; off >>= 1)
      acc[e] += __shfl_xor(acc[e], off, 32);
  if (lane == 0) {
    float m = acc[0];
#pragma unroll
    for (int e = 1; e < E_; ++e) m = fmaxf(m, acc[e]);
    float p[E_], s = 0.f;
#pragma unroll
    for (int e = 0; e < E_; ++e) { p[e] = __expf(acc[e] - m); s += p[e]; }
    float inv = 1.f / s;
    int b = t / S_, ss = t % S_;
#pragma unroll
    for (int e = 0; e < E_; ++e) Sm[((size_t)b * E_ + e) * S_ + ss] = p[e] * inv;
  }
}

// per (b,e): descending bitonic sort of 2048 scores in LDS, keep top 512
__global__ __launch_bounds__(512) void topk_bitonic(const float* __restrict__ Sm,
                                                    float* __restrict__ G,
                                                    int* __restrict__ I) {
  __shared__ float skey[S_];
  __shared__ int   sidx[S_];
  const int tid = threadIdx.x;
  const int pair = blockIdx.x;  // b*E+e
  const float* sp = Sm + (size_t)pair * S_;
  for (int i = tid; i < S_; i += 512) { skey[i] = sp[i]; sidx[i] = i; }
  __syncthreads();
  for (int k = 2; k <= S_; k <<= 1) {
    for (int j = k >> 1; j > 0; j >>= 1) {
      for (int t = tid; t < S_ / 2; t += 512) {
        int i = ((t & ~(j - 1)) << 1) | (t & (j - 1));
        int p = i | j;
        bool desc = ((i & k) == 0);
        float ki = skey[i], kp = skey[p];
        bool sw = desc ? (ki < kp) : (ki > kp);
        if (sw) {
          skey[i] = kp; skey[p] = ki;
          int ti = sidx[i]; sidx[i] = sidx[p]; sidx[p] = ti;
        }
      }
      __syncthreads();
    }
  }
  G[(size_t)pair * K_ + tid] = skey[tid];
  I[(size_t)pair * K_ + tid] = sidx[tid];
}

// ex[b,e,k,:] = bf16(x[b, I[b,e,k], :])
__global__ __launch_bounds__(256) void gather_cast(const float* __restrict__ x,
                                                   const int* __restrict__ I,
                                                   bf16* __restrict__ ex) {
  size_t t = ((size_t)blockIdx.x * 256 + threadIdx.x) * 4;
  const size_t perPair = (size_t)K_ * H_;
  int pair = (int)(t / perPair);
  size_t rem = t % perPair;
  int k = (int)(rem / H_), h = (int)(rem % H_);
  int b = pair >> 3;
  int tok = I[(size_t)pair * K_ + k];
  const float4 xv = *(const float4*)(x + ((size_t)b * S_ + tok) * H_ + h);
  union { bf16 h4[4]; uint2 u; } o;
  o.h4[0] = (bf16)xv.x; o.h4[1] = (bf16)xv.y; o.h4[2] = (bf16)xv.z; o.h4[3] = (bf16)xv.w;
  *(uint2*)(ex + t) = o.u;
}

// h1 = gelu(ex @ W1 + b1)   [per (b,e): 512x1024 * 1024x4096]
__global__ __launch_bounds__(256) void gemm1_gelu(const bf16* __restrict__ ex,
                                                  const bf16* __restrict__ W1T,
                                                  const float* __restrict__ b1,
                                                  bf16* __restrict__ h1) {
  __shared__ bf16 sA[2][BM * LDT];
  __shared__ bf16 sB[2][BN * LDT];
  const int tid = threadIdx.x;
  const int pair = blockIdx.z, e = pair & (E_ - 1);
  const bf16* Ag = ex + (size_t)pair * K_ * H_ + (size_t)blockIdx.y * BM * H_;
  const bf16* Bg = W1T + (size_t)e * F_ * H_ + (size_t)blockIdx.x * BN * H_;
  FragC acc[2][4];
#pragma unroll
  for (int i = 0; i < 2; ++i)
#pragma unroll
    for (int j = 0; j < 4; ++j)
#pragma unroll
      for (int r = 0; r < 8; ++r) acc[i][j].f[r] = 0.f;
  gemm_tile(Ag, H_, Bg, H_, H_, tid, sA, sB, acc);

  const int lane = tid & 31, wave = tid >> 5;
  const int wm = (wave >> 1) * 32, wn = (wave & 1) * 64;
  const int grp = lane >> 4, nl = lane & 15;
  bf16* h1p = h1 + (size_t)pair * K_ * F_;
  const float* b1e = b1 + (size_t)e * F_;
#pragma unroll
  for (int i = 0; i < 2; ++i) {
    int r0 = blockIdx.y * BM + wm + i * 16 + grp * 8;
#pragma unroll
    for (int j = 0; j < 4; ++j) {
      int c = blockIdx.x * BN + wn + j * 16 + nl;
      float bias = b1e[c];
#pragma unroll
      for (int r = 0; r < 8; ++r) {
        float v = acc[i][j].f[r] + bias;
        float g = 0.5f * v * (1.0f + tanhf(0.7978845608028654f * (v + 0.044715f * v * v * v)));
        h1p[(size_t)(r0 + r) * F_ + c] = (bf16)g;
      }
    }
  }
}

// out += scatter( G * (h1 @ W2 + b2) )   [per (b,e): 512x4096 * 4096x1024]
__global__ __launch_bounds__(256) void gemm2_combine(const bf16* __restrict__ h1,
                                                     const bf16* __restrict__ W2T,
                                                     const float* __restrict__ b2,
                                                     const float* __restrict__ G,
                                                     const int* __restrict__ I,
                                                     float* __restrict__ out) {
  __shared__ bf16 sA[2][BM * LDT];
  __shared__ bf16 sB[2][BN * LDT];
  const int tid = threadIdx.x;
  const int pair = blockIdx.z, b = pair >> 3, e = pair & 7;
  const bf16* Ag = h1 + (size_t)pair * K_ * F_ + (size_t)blockIdx.y * BM * F_;
  const bf16* Bg = W2T + (size_t)e * H_ * F_ + (size_t)blockIdx.x * BN * F_;
  FragC acc[2][4];
#pragma unroll
  for (int i = 0; i < 2; ++i)
#pragma unroll
    for (int j = 0; j < 4; ++j)
#pragma unroll
      for (int r = 0; r < 8; ++r) acc[i][j].f[r] = 0.f;
  gemm_tile(Ag, F_, Bg, F_, F_, tid, sA, sB, acc);

  const int lane = tid & 31, wave = tid >> 5;
  const int wm = (wave >> 1) * 32, wn = (wave & 1) * 64;
  const int grp = lane >> 4, nl = lane & 15;
  const float* Gp = G + (size_t)pair * K_;
  const int*   Ip = I + (size_t)pair * K_;
  const float* b2e = b2 + (size_t)e * H_;
#pragma unroll
  for (int i = 0; i < 2; ++i) {
    int r0 = blockIdx.y * BM + wm + i * 16 + grp * 8;
    float gate[8]; int tok[8];
#pragma unroll
    for (int r = 0; r < 8; ++r) { gate[r] = Gp[r0 + r]; tok[r] = Ip[r0 + r]; }
#pragma unroll
    for (int j = 0; j < 4; ++j) {
      int c = blockIdx.x * BN + wn + j * 16 + nl;
      float bias = b2e[c];
#pragma unroll
      for (int r = 0; r < 8; ++r) {
        float v = (acc[i][j].f[r] + bias) * gate[r];
        atomicAdd(out + ((size_t)(b * S_ + tok[r])) * H_ + c, v);
      }
    }
  }
}

// ---------------- launch ----------------
extern "C" void kernel_launch(void* const* d_in, const int* in_sizes, int n_in,
                              void* d_out, int out_size, void* d_ws, size_t ws_size,
                              hipStream_t stream) {
  (void)in_sizes; (void)n_in; (void)out_size; (void)ws_size;
  const float* x  = (const float*)d_in[0];
  const float* Wg = (const float*)d_in[1];
  const float* W1 = (const float*)d_in[2];
  const float* b1 = (const float*)d_in[3];
  const float* W2 = (const float*)d_in[4];
  const float* b2 = (const float*)d_in[5];
  float* out = (float*)d_out;

  char* w = (char*)d_ws;
  size_t off = 0;
  auto alloc = [&](size_t bytes) {
    off = (off + 255) & ~(size_t)255;
    void* p = w + off; off += bytes; return p;
  };
  float* Sm  = (float*)alloc((size_t)B_ * E_ * S_ * 4);        // 256 KB
  float* G   = (float*)alloc((size_t)B_ * E_ * K_ * 4);        // 64 KB
  int*   I   = (int*)  alloc((size_t)B_ * E_ * K_ * 4);        // 64 KB
  bf16*  ex  = (bf16*) alloc((size_t)B_ * E_ * K_ * H_ * 2);   // 32 MB
  bf16*  W1T = (bf16*) alloc((size_t)E_ * F_ * H_ * 2);        // 64 MB
  bf16*  W2T = (bf16*) alloc((size_t)E_ * H_ * F_ * 2);        // 64 MB
  bf16*  h1  = (bf16*) alloc((size_t)B_ * E_ * K_ * F_ * 2);   // 128 MB

  zero4<<<(B_ * S_ * H_ / 4) / 256, 256, 0, stream>>>(out);
  conv_w1t<<<(int)(((size_t)E_ * F_ * H_) / 256), 256, 0, stream>>>(W1, W1T);
  conv_w2t<<<(int)(((size_t)E_ * H_ * F_) / 256), 256, 0, stream>>>(W2, W2T);
  gate_softmax<<<B_ * S_ / 8, 256, 0, stream>>>(x, Wg, Sm);
  topk_bitonic<<<B_ * E_, 512, 0, stream>>>(Sm, G, I);
  gather_cast<<<(int)(((size_t)B_ * E_ * K_ * H_ / 4) / 256), 256, 0, stream>>>(x, I, ex);
  gemm1_gelu<<<dim3(F_ / BN, K_ / BM, B_ * E_), 256, 0, stream>>>(ex, W1T, b1, h1);
  gemm2_combine<<<dim3(H_ / BN, K_ / BM, B_ * E_), 256, 0, stream>>>(h1, W2T, b2, G, I, out);
}
